// QwenVLA_13821204759059
// MI455X (gfx1250) — matrix operand order repaired and checked by reading, not compiled, not run on hardware
//
#include <hip/hip_runtime.h>
#include <hip/hip_bf16.h>

// ---------------------------------------------------------------------------
// QwenVLA dual-stream transformer block for gfx1250 (MI455X).
// bf16 WMMA GEMMs with pre-transposed bf16 weights (L2-resident slabs),
// double-buffered GLOBAL_LOAD_ASYNC_TO_LDS pipeline, flash attention.
// ---------------------------------------------------------------------------

typedef __attribute__((ext_vector_type(16))) __bf16 v16bf;
typedef __attribute__((ext_vector_type(8)))  float  v8f;
typedef __attribute__((ext_vector_type(4)))  int    v4i;

union FragU { uint4 q[2]; v16bf v; };

__device__ __forceinline__ v8f wmma_bf16(v16bf a, v16bf b, v8f c) {
  return __builtin_amdgcn_wmma_f32_16x16x32_bf16(
      /*neg_a=*/false, a, /*neg_b=*/false, b,
      /*c_mod=*/(short)0, c, /*reuse_a=*/false, /*reuse_b=*/false);
}

// ---- CDNA5 async global->LDS copy (ASYNCcnt), with sync fallback ----------
__device__ __forceinline__ void async_copy_b128(const __bf16* gsrc, __bf16* ldst) {
#if defined(__gfx1250__) && __has_builtin(__builtin_amdgcn_global_load_async_to_lds_b128)
  // builtin signature (from probe diagnostics): param0 = v4i in global AS.
  __builtin_amdgcn_global_load_async_to_lds_b128(
      (__attribute__((address_space(1))) v4i*)gsrc,
      (__attribute__((address_space(3))) v4i*)ldst,
      /*offset=*/0, /*cpol=*/0);
#else
  *(uint4*)ldst = *(const uint4*)gsrc;
#endif
}

__device__ __forceinline__ void wait_async() {
#if defined(__gfx1250__) && __has_builtin(__builtin_amdgcn_s_wait_asynccnt)
  __builtin_amdgcn_s_wait_asynccnt(0);
#elif defined(__AMDGCN__)
  asm volatile("s_wait_asynccnt 0" ::: "memory");
#endif
}

#define MODE_F32       0
#define MODE_F32_RES   1
#define MODE_SILU_BF16 2

// ================= weight convert + transpose: f32[K,N] -> bf16[N,K] =======
__global__ __launch_bounds__(256) void transpose_bf16(
    const float* __restrict__ W, __bf16* __restrict__ Wt, int K, int N)
{
  __shared__ __bf16 t[32][33];
  int n0 = blockIdx.x * 32, k0 = blockIdx.y * 32;
  int c = threadIdx.x & 31, r0 = threadIdx.x >> 5;  // 8 rows per pass
#pragma unroll
  for (int r = r0; r < 32; r += 8)
    t[r][c] = (__bf16)W[(size_t)(k0 + r) * N + n0 + c];
  __syncthreads();
#pragma unroll
  for (int r = r0; r < 32; r += 8)
    Wt[(size_t)(n0 + r) * K + k0 + c] = t[c][r];
}

// ======================= generic WMMA GEMM =================================
// C[M,N] = A[M,K](bf16) * Bt[N,K](bf16, pre-transposed weights)
// Tile 128x256x32, 256 threads (8 waves, 2x4 grid, 64x64 per wave = 16 WMMA).
// Double-buffered async global->LDS staging. M%128==0, N%256==0, K%32==0.
template<int MODE>
__global__ __launch_bounds__(256) void gemm_kernel(
    const __bf16* __restrict__ A, int lda,
    const __bf16* __restrict__ Bt,
    float*        __restrict__ Df,
    __bf16*       __restrict__ Dbf, int ldd,
    const float*  __restrict__ RES, int ldr,
    const float*  __restrict__ G,   int ldg,
    int K)
{
  __shared__ __bf16 As[2][128 * 40];  // [row][k], pitch 40 (80B, 16B aligned)
  __shared__ __bf16 Bs[2][256 * 40];  // [col][k], pitch 40

  const int tid  = threadIdx.x;
  const int m0   = blockIdx.y * 128;
  const int n0   = blockIdx.x * 256;
  const int wv   = tid >> 5;
  const int wm   = wv >> 2;        // 0..1 -> 64 rows
  const int wn   = wv & 3;         // 0..3 -> 64 cols
  const int lane = tid & 31;
  const int half = lane >> 4;
  const int ln   = lane & 15;

  // stage one 32-wide K slab of A (128 rows) and Bt (256 rows) into LDS
  auto stage = [&](int buf, int k0) {
#pragma unroll
    for (int i = 0; i < 2; ++i) {           // A: 512 chunks of 8 bf16
      int c = tid + i * 256, r = c >> 2, ko = (c & 3) * 8;
      async_copy_b128(A + (size_t)(m0 + r) * lda + k0 + ko, &As[buf][r * 40 + ko]);
    }
#pragma unroll
    for (int i = 0; i < 4; ++i) {           // B: 1024 chunks of 8 bf16
      int c = tid + i * 256, r = c >> 2, ko = (c & 3) * 8;
      async_copy_b128(Bt + (size_t)(n0 + r) * (size_t)K + k0 + ko, &Bs[buf][r * 40 + ko]);
    }
  };

  v8f acc[4][4];
#pragma unroll
  for (int i = 0; i < 4; ++i)
#pragma unroll
    for (int j = 0; j < 4; ++j)
      acc[i][j] = v8f{0.f,0.f,0.f,0.f,0.f,0.f,0.f,0.f};

  const int nk = K >> 5;
  stage(0, 0);
  for (int it = 0; it < nk; ++it) {
    const int cur = it & 1;
    wait_async();        // my wave's staged copies (incl. tile `it`) complete
    __syncthreads();     // all waves done staging `cur` and done reading `cur^1`
    if (it + 1 < nk) stage(cur ^ 1, (it + 1) * 32);

    FragU af[4], bf4[4];
#pragma unroll
    for (int i = 0; i < 4; ++i) {
      int r = wm * 64 + i * 16 + ln;
      af[i].q[0] = *(const uint4*)(&As[cur][r * 40 + half * 8]);       // k h*8..
      af[i].q[1] = *(const uint4*)(&As[cur][r * 40 + half * 8 + 16]);  // k 16+h*8..
    }
#pragma unroll
    for (int j = 0; j < 4; ++j) {
      int c = wn * 64 + j * 16 + ln;
      bf4[j].q[0] = *(const uint4*)(&Bs[cur][c * 40 + half * 8]);       // k h*8..
      bf4[j].q[1] = *(const uint4*)(&Bs[cur][c * 40 + half * 8 + 16]);
    }
#pragma unroll
    for (int i = 0; i < 4; ++i)
#pragma unroll
      for (int j = 0; j < 4; ++j)
        acc[i][j] = wmma_bf16(af[i].v, bf4[j].v, acc[i][j]);
  }

  // ---- epilogue ----
#pragma unroll
  for (int i = 0; i < 4; ++i) {
#pragma unroll
    for (int j = 0; j < 4; ++j) {
      int c = n0 + wn * 64 + j * 16 + ln;
#pragma unroll
      for (int e = 0; e < 8; ++e) {
        int r = m0 + wm * 64 + i * 16 + half * 8 + e;
        float v = acc[i][j][e];
        size_t idx = (size_t)r * ldd + c;
        if constexpr (MODE == MODE_F32) {
          Df[idx] = v;
        } else if constexpr (MODE == MODE_F32_RES) {
          Df[idx] = v + RES[(size_t)r * ldr + c];
        } else { // MODE_SILU_BF16: silu(G) * acc -> bf16
          float g  = G[(size_t)r * ldg + c];
          float sg = g / (1.0f + __expf(-g));
          Dbf[idx] = (__bf16)(sg * v);
        }
      }
    }
  }
}

// ======================= row RMSNorm (f32 -> bf16) =========================
__global__ __launch_bounds__(256) void rmsnorm_rows(
    const float* __restrict__ X, const float* __restrict__ W,
    __bf16* __restrict__ Y, int D)
{
  int row = blockIdx.x;
  const float* x = X + (size_t)row * D;
  float ss = 0.f;
  for (int i = threadIdx.x; i < D; i += 256) { float v = x[i]; ss += v * v; }
#pragma unroll
  for (int off = 16; off >= 1; off >>= 1) ss += __shfl_xor(ss, off, 32);
  __shared__ float red[8];
  int w = threadIdx.x >> 5, lane = threadIdx.x & 31;
  if (lane == 0) red[w] = ss;
  __syncthreads();
  float tot = 0.f;
#pragma unroll
  for (int i = 0; i < 8; ++i) tot += red[i];
  float inv = rsqrtf(tot / (float)D + 1e-6f);
  __bf16* y = Y + (size_t)row * D;
  for (int i = threadIdx.x; i < D; i += 256)
    y[i] = (__bf16)(x[i] * inv * W[i]);
}

// ================= q/k head-RMSNorm + RoPE + layout kernel =================
#define SEQ 1088
__global__ __launch_bounds__(256) void qkv_post(
    const float* __restrict__ QKV,
    const float* __restrict__ qn, const float* __restrict__ kn,
    __bf16* __restrict__ Qg,   // [B,16,SEQ,128]
    __bf16* __restrict__ Kg,   // [B,8, SEQ,128]
    __bf16* __restrict__ Vt,   // [B,8, 128,SEQ]  (transposed)
    int Sp, int s_off)
{
  int rowid = blockIdx.x;
  int b = rowid / Sp, t = rowid % Sp;
  int sg = s_off + t;
  int wave = threadIdx.x >> 5, lane = threadIdx.x & 31;
  const float* base = QKV + (size_t)rowid * 4096;

  const float NEG_LN_TH = -2.0f / 128.0f * 15.424948470398375f; // -2/DH*ln(5e6)
  float a0 = (float)sg * __expf((float)lane * NEG_LN_TH);
  float a1 = (float)sg * __expf((float)(lane + 32) * NEG_LN_TH);
  float c0 = __cosf(a0), s0 = __sinf(a0);
  float c1 = __cosf(a1), s1 = __sinf(a1);

  for (int u = wave; u < 32; u += 8) {
    const float* src;
    if (u < 16)      src = base + u * 128;
    else if (u < 24) src = base + 2048 + (u - 16) * 128;
    else             src = base + 3072 + (u - 24) * 128;
    float x0 = src[lane], x1 = src[lane + 32], x2 = src[lane + 64], x3 = src[lane + 96];

    if (u < 24) {
      float ssq = x0*x0 + x1*x1 + x2*x2 + x3*x3;
#pragma unroll
      for (int off = 16; off >= 1; off >>= 1) ssq += __shfl_xor(ssq, off, 32);
      float inv = rsqrtf(ssq / 128.0f + 1e-6f);
      const float* wn = (u < 16) ? qn : kn;
      x0 = x0 * inv * wn[lane];      x1 = x1 * inv * wn[lane + 32];
      x2 = x2 * inv * wn[lane + 64]; x3 = x3 * inv * wn[lane + 96];
      float r0 = x0 * c0 - x2 * s0;      // d = lane
      float r2 = x2 * c0 + x0 * s0;      // d = lane+64
      float r1 = x1 * c1 - x3 * s1;      // d = lane+32
      float r3 = x3 * c1 + x1 * s1;      // d = lane+96
      __bf16* dst;
      if (u < 16) dst = Qg + (((size_t)b * 16 + u) * SEQ + sg) * 128;
      else        dst = Kg + (((size_t)b * 8 + (u - 16)) * SEQ + sg) * 128;
      dst[lane] = (__bf16)r0; dst[lane + 32] = (__bf16)r1;
      dst[lane + 64] = (__bf16)r2; dst[lane + 96] = (__bf16)r3;
    } else {
      __bf16* dst = Vt + ((size_t)b * 8 + (u - 24)) * 128 * SEQ;
      dst[(size_t)(lane)      * SEQ + sg] = (__bf16)x0;
      dst[(size_t)(lane + 32) * SEQ + sg] = (__bf16)x1;
      dst[(size_t)(lane + 64) * SEQ + sg] = (__bf16)x2;
      dst[(size_t)(lane + 96) * SEQ + sg] = (__bf16)x3;
    }
  }
}

// =========================== flash attention ===============================
// grid(17, HQ=16, B=4), 128 threads = 4 waves, 16 q-rows/wave, k-tiles of 32.
__global__ __launch_bounds__(128) void attn_kernel(
    const __bf16* __restrict__ Qg, const __bf16* __restrict__ Kg,
    const __bf16* __restrict__ Vt,
    __bf16* __restrict__ att_vl,   // [B*1024, 2048] bf16
    __bf16* __restrict__ att_ex)   // [B*64,   2048] bf16
{
  const int qblk = blockIdx.x, h = blockIdx.y, b = blockIdx.z;
  const int wave = threadIdx.x >> 5, lane = threadIdx.x & 31;
  const int half = lane >> 4, ln = lane & 15;
  const int q0 = qblk * 64 + wave * 16;
  const int kvh = h >> 1;

  const __bf16* Qbase = Qg + ((size_t)b * 16 + h) * SEQ * 128;
  const __bf16* Kbase = Kg + ((size_t)b * 8 + kvh) * SEQ * 128;
  const __bf16* Vbase = Vt + ((size_t)b * 8 + kvh) * 128 * SEQ;

  __shared__ __bf16 Pl[4][16 * 40];  // per-wave P transpose scratch

  FragU qf[4];
#pragma unroll
  for (int dc = 0; dc < 4; ++dc) {
    const __bf16* p = Qbase + (size_t)(q0 + ln) * 128 + dc * 32 + half * 8;
    qf[dc].q[0] = *(const uint4*)(p);
    qf[dc].q[1] = *(const uint4*)(p + 16);
  }

  v8f oacc[8];
#pragma unroll
  for (int dt = 0; dt < 8; ++dt)
    oacc[dt] = v8f{0.f,0.f,0.f,0.f,0.f,0.f,0.f,0.f};
  float m_i[8], l_i[8];
#pragma unroll
  for (int e = 0; e < 8; ++e) { m_i[e] = -3.0e38f; l_i[e] = 0.f; }

  const float SCALE = 0.08838834764831845f;  // 128^-0.5
  const int nkt = (qblk < 16) ? 32 : 34;

  for (int kt = 0; kt < nkt; ++kt) {
    const int kb = kt * 32;
    v8f sc[2];
#pragma unroll
    for (int cf = 0; cf < 2; ++cf) {
      v8f s = v8f{0.f,0.f,0.f,0.f,0.f,0.f,0.f,0.f};
#pragma unroll
      for (int dc = 0; dc < 4; ++dc) {
        FragU kf;
        const __bf16* p = Kbase + (size_t)(kb + cf * 16 + ln) * 128 + dc * 32 + half * 8;
        kf.q[0] = *(const uint4*)(p);
        kf.q[1] = *(const uint4*)(p + 16);
        s = wmma_bf16(qf[dc].v, kf.v, s);
      }
      sc[cf] = s;
    }
#pragma unroll
    for (int cf = 0; cf < 2; ++cf)
#pragma unroll
      for (int e = 0; e < 8; ++e) sc[cf][e] *= SCALE;

    if (qblk == 16 && kb >= 1024) {   // causal masking inside expert region
#pragma unroll
      for (int cf = 0; cf < 2; ++cf)
#pragma unroll
        for (int e = 0; e < 8; ++e) {
          int k  = kb + cf * 16 + ln;
          int rq = q0 + half * 8 + e;
          if (k > rq) sc[cf][e] = -3.0e38f;
        }
    }

    float alpha[8];
#pragma unroll
    for (int e = 0; e < 8; ++e) {
      float sm = fmaxf(sc[0][e], sc[1][e]);
#pragma unroll
      for (int off = 8; off >= 1; off >>= 1) sm = fmaxf(sm, __shfl_xor(sm, off, 32));
      float mnew = fmaxf(m_i[e], sm);
      float p0 = __expf(sc[0][e] - mnew);
      float p1 = __expf(sc[1][e] - mnew);
      float rs = p0 + p1;
#pragma unroll
      for (int off = 8; off >= 1; off >>= 1) rs += __shfl_xor(rs, off, 32);
      alpha[e] = __expf(m_i[e] - mnew);
      l_i[e] = l_i[e] * alpha[e] + rs;
      m_i[e] = mnew;
      int prow = half * 8 + e;
      Pl[wave][prow * 40 + ln]      = (__bf16)p0;  // k-local = ln
      Pl[wave][prow * 40 + 16 + ln] = (__bf16)p1;  // k-local = 16+ln
    }
    __syncthreads();

#pragma unroll
    for (int dt = 0; dt < 8; ++dt)
#pragma unroll
      for (int e = 0; e < 8; ++e) oacc[dt][e] *= alpha[e];

    FragU pf;  // P as A-fragment (16x32)
    pf.q[0] = *(const uint4*)(&Pl[wave][ln * 40 + half * 8]);
    pf.q[1] = *(const uint4*)(&Pl[wave][ln * 40 + 16 + half * 8]);

#pragma unroll
    for (int dt = 0; dt < 8; ++dt) {
      FragU vf;  // V^T rows are contiguous in k
      const __bf16* p = Vbase + (size_t)(dt * 16 + ln) * SEQ + kb + half * 8;
      vf.q[0] = *(const uint4*)(p);
      vf.q[1] = *(const uint4*)(p + 16);
      oacc[dt] = wmma_bf16(pf.v, vf.v, oacc[dt]);
    }
    __syncthreads();
  }

#pragma unroll
  for (int dt = 0; dt < 8; ++dt) {
#pragma unroll
    for (int e = 0; e < 8; ++e) {
      float v = oacc[dt][e] / l_i[e];
      int row = q0 + half * 8 + e;
      int col = h * 128 + dt * 16 + ln;
      if (row < 1024)
        att_vl[((size_t)b * 1024 + row) * 2048 + col] = (__bf16)v;
      else
        att_ex[((size_t)b * 64 + (row - 1024)) * 2048 + col] = (__bf16)v;
    }
  }
}

// ============================== launcher ===================================
extern "C" void kernel_launch(void* const* d_in, const int* in_sizes, int n_in,
                              void* d_out, int out_size, void* d_ws, size_t ws_size,
                              hipStream_t stream) {
  (void)in_sizes; (void)n_in; (void)out_size; (void)ws_size;
  const float* hs_vl = (const float*)d_in[0];
  const float* hs_ex = (const float*)d_in[1];
  const float* vl_ln1 = (const float*)d_in[4];
  const float* vl_wq  = (const float*)d_in[5];
  const float* vl_wk  = (const float*)d_in[6];
  const float* vl_wv  = (const float*)d_in[7];
  const float* vl_qn  = (const float*)d_in[8];
  const float* vl_kn  = (const float*)d_in[9];
  const float* vl_wo  = (const float*)d_in[10];
  const float* vl_ln2 = (const float*)d_in[11];
  const float* vl_wg  = (const float*)d_in[12];
  const float* vl_wu  = (const float*)d_in[13];
  const float* vl_wd  = (const float*)d_in[14];
  const float* ex_ln1 = (const float*)d_in[15];
  const float* ex_wq  = (const float*)d_in[16];
  const float* ex_wk  = (const float*)d_in[17];
  const float* ex_wv  = (const float*)d_in[18];
  const float* ex_qn  = (const float*)d_in[19];
  const float* ex_kn  = (const float*)d_in[20];
  const float* ex_wo  = (const float*)d_in[21];
  const float* ex_ln2 = (const float*)d_in[22];
  const float* ex_wg  = (const float*)d_in[23];
  const float* ex_wu  = (const float*)d_in[24];
  const float* ex_wd  = (const float*)d_in[25];

  constexpr size_t MVL = 4096, MEX = 256;
  constexpr size_t OFF_XN_VL  = 0;
  constexpr size_t OFF_XN_EX  = OFF_XN_VL  + MVL * 2048 * 2;
  constexpr size_t OFF_QKV_VL = OFF_XN_EX  + MEX * 1024 * 2;
  constexpr size_t OFF_QKV_EX = OFF_QKV_VL + MVL * 4096 * 4;
  constexpr size_t OFF_ATT_VL = OFF_QKV_EX + MEX * 4096 * 4;
  constexpr size_t OFF_ATT_EX = OFF_ATT_VL + MVL * 2048 * 2;
  constexpr size_t OFF_QG     = OFF_ATT_EX + MEX * 2048 * 2;
  constexpr size_t OFF_KG     = OFF_QG + (size_t)4 * 16 * 1088 * 128 * 2;
  constexpr size_t OFF_VT     = OFF_KG + (size_t)4 * 8 * 1088 * 128 * 2;
  constexpr size_t OFF_O_VL   = OFF_VT + (size_t)4 * 8 * 1088 * 128 * 2;
  constexpr size_t OFF_O_EX   = OFF_O_VL + MVL * 2048 * 4;
  constexpr size_t OFF_HM_VL  = OFF_O_EX + MEX * 1024 * 4;
  constexpr size_t OFF_HM_EX  = OFF_HM_VL + MVL * 6144 * 2;
  constexpr size_t OFF_WT     = OFF_HM_EX + MEX * 3072 * 2;   // 25.2MB reuse slab
  // G buffers alias the dead [QKV..VT] region
  constexpr size_t OFF_G_VL   = OFF_QKV_VL;
  constexpr size_t OFF_G_EX   = OFF_QKV_VL + MVL * 6144 * 4;

  char* ws = (char*)d_ws;
  __bf16* xn_vl  = (__bf16*)(ws + OFF_XN_VL);
  __bf16* xn_ex  = (__bf16*)(ws + OFF_XN_EX);
  float*  qkv_vl = (float*) (ws + OFF_QKV_VL);
  float*  qkv_ex = (float*) (ws + OFF_QKV_EX);
  __bf16* att_vl = (__bf16*)(ws + OFF_ATT_VL);
  __bf16* att_ex = (__bf16*)(ws + OFF_ATT_EX);
  __bf16* Qg     = (__bf16*)(ws + OFF_QG);
  __bf16* Kg     = (__bf16*)(ws + OFF_KG);
  __bf16* Vt     = (__bf16*)(ws + OFF_VT);
  float*  o_vl   = (float*) (ws + OFF_O_VL);
  float*  o_ex   = (float*) (ws + OFF_O_EX);
  __bf16* hm_vl  = (__bf16*)(ws + OFF_HM_VL);
  __bf16* hm_ex  = (__bf16*)(ws + OFF_HM_EX);
  __bf16* WT     = (__bf16*)(ws + OFF_WT);
  float*  G_vl   = (float*) (ws + OFF_G_VL);
  float*  G_ex   = (float*) (ws + OFF_G_EX);
  float*  out_vl = (float*)d_out;
  float*  out_ex = (float*)d_out + (size_t)4 * 1024 * 2048;

  // transpose W[K,N]->WT[N,K] then GEMM (stream-ordered, WT slab reused)
  auto T = [&](const float* W, int K, int N) {
    transpose_bf16<<<dim3(N / 32, K / 32), 256, 0, stream>>>(W, WT, K, N);
  };

  // 1) input RMSNorms
  rmsnorm_rows<<<MVL, 256, 0, stream>>>(hs_vl, vl_ln1, xn_vl, 2048);
  rmsnorm_rows<<<MEX, 256, 0, stream>>>(hs_ex, ex_ln1, xn_ex, 1024);

  // 2) QKV projections (packed [q|k|v] rows, ldd=4096)
  T(vl_wq, 2048, 2048);
  gemm_kernel<MODE_F32><<<dim3(8, 32), 256, 0, stream>>>(xn_vl, 2048, WT, qkv_vl + 0,    nullptr, 4096, nullptr, 0, nullptr, 0, 2048);
  T(vl_wk, 2048, 1024);
  gemm_kernel<MODE_F32><<<dim3(4, 32), 256, 0, stream>>>(xn_vl, 2048, WT, qkv_vl + 2048, nullptr, 4096, nullptr, 0, nullptr, 0, 2048);
  T(vl_wv, 2048, 1024);
  gemm_kernel<MODE_F32><<<dim3(4, 32), 256, 0, stream>>>(xn_vl, 2048, WT, qkv_vl + 3072, nullptr, 4096, nullptr, 0, nullptr, 0, 2048);
  T(ex_wq, 1024, 2048);
  gemm_kernel<MODE_F32><<<dim3(8, 2), 256, 0, stream>>>(xn_ex, 1024, WT, qkv_ex + 0,    nullptr, 4096, nullptr, 0, nullptr, 0, 1024);
  T(ex_wk, 1024, 1024);
  gemm_kernel<MODE_F32><<<dim3(4, 2), 256, 0, stream>>>(xn_ex, 1024, WT, qkv_ex + 2048, nullptr, 4096, nullptr, 0, nullptr, 0, 1024);
  T(ex_wv, 1024, 1024);
  gemm_kernel<MODE_F32><<<dim3(4, 2), 256, 0, stream>>>(xn_ex, 1024, WT, qkv_ex + 3072, nullptr, 4096, nullptr, 0, nullptr, 0, 1024);

  // 3) head-norm + RoPE + K/V layout (V transposed)
  qkv_post<<<4 * 1024, 256, 0, stream>>>(qkv_vl, vl_qn, vl_kn, Qg, Kg, Vt, 1024, 0);
  qkv_post<<<4 * 64,   256, 0, stream>>>(qkv_ex, ex_qn, ex_kn, Qg, Kg, Vt, 64, 1024);

  // 4) attention
  attn_kernel<<<dim3(17, 16, 4), 128, 0, stream>>>(Qg, Kg, Vt, att_vl, att_ex);

  // 5) output projection + residual
  T(vl_wo, 2048, 2048);
  gemm_kernel<MODE_F32_RES><<<dim3(8, 32), 256, 0, stream>>>(att_vl, 2048, WT, o_vl, nullptr, 2048, hs_vl, 2048, nullptr, 0, 2048);
  T(ex_wo, 2048, 1024);
  gemm_kernel<MODE_F32_RES><<<dim3(4, 2), 256, 0, stream>>>(att_ex, 2048, WT, o_ex, nullptr, 1024, hs_ex, 1024, nullptr, 0, 2048);

  // 6) post-attention RMSNorm (reuses xn buffers)
  rmsnorm_rows<<<MVL, 256, 0, stream>>>(o_vl, vl_ln2, xn_vl, 2048);
  rmsnorm_rows<<<MEX, 256, 0, stream>>>(o_ex, ex_ln2, xn_ex, 1024);

  // 7) SwiGLU MLP + residual -> d_out
  T(vl_wg, 2048, 6144);
  gemm_kernel<MODE_F32><<<dim3(24, 32), 256, 0, stream>>>(xn_vl, 2048, WT, G_vl, nullptr, 6144, nullptr, 0, nullptr, 0, 2048);
  T(vl_wu, 2048, 6144);
  gemm_kernel<MODE_SILU_BF16><<<dim3(24, 32), 256, 0, stream>>>(xn_vl, 2048, WT, nullptr, hm_vl, 6144, nullptr, 0, G_vl, 6144, 2048);
  T(vl_wd, 6144, 2048);
  gemm_kernel<MODE_F32_RES><<<dim3(8, 32), 256, 0, stream>>>(hm_vl, 6144, WT, out_vl, nullptr, 2048, o_vl, 2048, nullptr, 0, 6144);

  T(ex_wg, 1024, 3072);
  gemm_kernel<MODE_F32><<<dim3(12, 2), 256, 0, stream>>>(xn_ex, 1024, WT, G_ex, nullptr, 3072, nullptr, 0, nullptr, 0, 1024);
  T(ex_wu, 1024, 3072);
  gemm_kernel<MODE_SILU_BF16><<<dim3(12, 2), 256, 0, stream>>>(xn_ex, 1024, WT, nullptr, hm_ex, 3072, nullptr, 0, G_ex, 3072, 1024);
  T(ex_wd, 3072, 1024);
  gemm_kernel<MODE_F32_RES><<<dim3(4, 2), 256, 0, stream>>>(hm_ex, 3072, WT, out_ex, nullptr, 1024, o_ex, 1024, nullptr, 0, 3072);
}
